// FastSpeech_51479478010589
// MI455X (gfx1250) — matrix operand order repaired and checked by reading, not compiled, not run on hardware
//
// FastSpeech forward for gfx1250 (MI455X): bf16 WMMA GEMM core (64x128 macro tile,
// 2x2 WMMA per wave, ping-pong LDS, vectorized staging) + fused aux kernels.
#include <hip/hip_runtime.h>
#include <math.h>

typedef __attribute__((ext_vector_type(16))) __bf16        v16bf;
typedef __attribute__((ext_vector_type(8)))  float         v8f;
typedef __attribute__((ext_vector_type(4)))  unsigned int  u32x4;
typedef __attribute__((ext_vector_type(2)))  unsigned int  u32x2;
typedef __attribute__((ext_vector_type(4)))  float         f32x4;
typedef long long ll;

namespace fs {
constexpr int B = 16, S = 128, D = 384, DF = 1536, H = 2, L = 3;
constexpr int NM = 80, LOUT = 1024, DH = D / H;            // DH = 192
constexpr int TM = 64, TN = 128, TK = 32;                  // 8 waves, each 32x32 (2x2 WMMA)
}

// ---------------------------------------------------------------- utilities
__device__ __forceinline__ unsigned short f2bf(float f) {
  union { float f; unsigned u; } v; v.f = f;
  unsigned r = (v.u + 0x7FFFu + ((v.u >> 16) & 1u)) >> 16;   // round-to-nearest-even
  return (unsigned short)r;
}
__device__ __forceinline__ unsigned pk2bf(float a, float b) {
  return (unsigned)f2bf(a) | ((unsigned)f2bf(b) << 16);
}
__device__ __forceinline__ float wsum(float v) {
  for (int m = 16; m; m >>= 1) v += __shfl_xor(v, m, 32);
  return v;
}
__device__ __forceinline__ float wmax(float v) {
  for (int m = 16; m; m >>= 1) v = fmaxf(v, __shfl_xor(v, m, 32));
  return v;
}

// ---------------------------------------------------------------- WMMA GEMM
// C[z][m][n] = act( alpha * sum_k A[z][m][k] * B[z][k][n] + bias[n] )
// flags: 1=ReLU, 2=B is [N,K] row-major (transposed), 4=A is im2col of a
// [B*convT, convCin] activation, 3-tap 'SAME' conv, Kd = 3*convCin.
// REQUIRES (guaranteed by call sites): M % TM == 0, Kd % TK == 0, convT power of 2.
struct GemmP {
  const float* A; const float* Bm; float* C; const float* bias;
  ll lda, ldb, ldc;
  ll sAb, sAh, sBb, sBh, sCb, sCh;   // z = zb*Hh + zh two-level batch strides
  int M, N, Kd, Hh;
  int flags;
  int convT, convTsh, convCin;
  float alpha;
};

__device__ __forceinline__ void stage_tile(const GemmP& p,
                                           const float* __restrict__ A,
                                           const float* __restrict__ Bm,
                                           unsigned short (*__restrict__ As)[fs::TK],
                                           unsigned short (*__restrict__ Bs)[fs::TK],
                                           int tileM, int tileN, int k0, int tid,
                                           bool im2c, bool trb) {
  using namespace fs;
  const bool pf = (k0 + TK < p.Kd);
  // ---- A tile: TM x TK, float4 groups along k (512 groups, 2/thread)
  #pragma unroll
  for (int i = 0; i < (TM * TK) / (256 * 4); ++i) {
    int g = tid + i * 256;
    int mr = g >> 3, kk = (g & 7) << 2;
    int m = tileM + mr, gk = k0 + kk;
    f32x4 v = {0.f, 0.f, 0.f, 0.f};
    if (im2c) {
      int b = m >> p.convTsh, t = m & (p.convT - 1);
      int kt = (gk >= 2 * p.convCin) ? 2 : ((gk >= p.convCin) ? 1 : 0);
      int ci = gk - kt * p.convCin;
      int ts = t + kt - 1;                              // 'SAME' pad, K=3
      if (ts >= 0 && ts < p.convT)
        v = *(const f32x4*)(A + (ll)((b << p.convTsh) + ts) * p.lda + ci);
    } else {
      const float* ap = A + (ll)m * p.lda + gk;
      v = *(const f32x4*)ap;
      if (pf) __builtin_prefetch(ap + TK, 0, 1);        // global_prefetch next k-slice
    }
    u32x2 w;
    w[0] = pk2bf(v[0], v[1]);
    w[1] = pk2bf(v[2], v[3]);
    *(u32x2*)(&As[mr][kk]) = w;
  }
  // ---- B tile: TK x TN stored transposed as Bs[n][k] (1024 groups, 4/thread)
  if (trb) {
    #pragma unroll
    for (int i = 0; i < (TN * TK) / (256 * 4); ++i) {
      int g = tid + i * 256;
      int n = g >> 3, kk = (g & 7) << 2;
      int gn = tileN + n, gk = k0 + kk;
      f32x4 v = {0.f, 0.f, 0.f, 0.f};
      if (gn < p.N) {
        const float* bp = Bm + (ll)gn * p.ldb + gk;
        v = *(const f32x4*)bp;
        if (pf) __builtin_prefetch(bp + TK, 0, 1);
      }
      u32x2 w;
      w[0] = pk2bf(v[0], v[1]);
      w[1] = pk2bf(v[2], v[3]);
      *(u32x2*)(&Bs[n][kk]) = w;
    }
  } else {
    #pragma unroll
    for (int i = 0; i < (TN * TK) / (256 * 4); ++i) {
      int g = tid + i * 256;
      int kk = g >> 5, n = (g & 31) << 2;
      int gn = tileN + n, gk = k0 + kk;
      f32x4 v = {0.f, 0.f, 0.f, 0.f};
      if (gn < p.N) {
        const float* bp = Bm + (ll)gk * p.ldb + gn;
        v = *(const f32x4*)bp;
        if (pf) __builtin_prefetch(bp + (ll)TK * p.ldb, 0, 1);
      }
      Bs[n + 0][kk] = f2bf(v[0]);
      Bs[n + 1][kk] = f2bf(v[1]);
      Bs[n + 2][kk] = f2bf(v[2]);
      Bs[n + 3][kk] = f2bf(v[3]);
    }
  }
}

__global__ __launch_bounds__(256) void k_gemm(GemmP p) {
  using namespace fs;
  __shared__ __align__(16) unsigned short As[2][TM][TK];   // ping-pong, bf16
  __shared__ __align__(16) unsigned short Bs[2][TN][TK];   // B transposed in LDS

  const int z  = blockIdx.z;
  const int zb = z / p.Hh, zh = z - zb * p.Hh;
  const float* A  = p.A  + (ll)zb * p.sAb + (ll)zh * p.sAh;
  const float* Bm = p.Bm + (ll)zb * p.sBb + (ll)zh * p.sBh;
  float*       C  = p.C  + (ll)zb * p.sCb + (ll)zh * p.sCh;

  const int tileM = blockIdx.y * TM;
  const int tileN = blockIdx.x * TN;
  const int tid   = threadIdx.x;
  const int lane  = tid & 31;
  const int wave  = tid >> 5;
  const int wr    = wave >> 2;        // 0..1  -> 32-row band
  const int wc    = wave & 3;         // 0..3  -> 32-col band
  const bool im2c = (p.flags & 4) != 0;
  const bool trb  = (p.flags & 2) != 0;

  v8f accs[2][2];
  #pragma unroll
  for (int si = 0; si < 2; ++si)
    #pragma unroll
    for (int sj = 0; sj < 2; ++sj)
      accs[si][sj] = (v8f){0.f, 0.f, 0.f, 0.f, 0.f, 0.f, 0.f, 0.f};

  stage_tile(p, A, Bm, As[0], Bs[0], tileM, tileN, 0, tid, im2c, trb);
  __syncthreads();

  const int nIter = p.Kd / TK;
  for (int it = 0; it < nIter; ++it) {
    const int cur = it & 1;
    const int kNext = (it + 1) * TK;
    if (kNext < p.Kd)
      stage_tile(p, A, Bm, As[cur ^ 1], Bs[cur ^ 1], tileM, tileN, kNext, tid, im2c, trb);

    // fragment gather per documented 16-bit WMMA VGPR layouts
    const u32x4* As4 = (const u32x4*)(&As[cur][0][0]);   // 1 u32x4 == 8 bf16
    const u32x4* Bs4 = (const u32x4*)(&Bs[cur][0][0]);
    const int m0 = wr * 32 + (lane & 15);
    const int n0 = wc * 32 + (lane & 15);
    const int hi = lane >> 4;
    union Frag { u32x4 q[2]; v16bf v; };
    Frag a0, a1, b0, b1;
    // A lane: row m; K = {8*hi+0..7, 16+8*hi+0..7}
    a0.q[0] = As4[m0 * 4 + hi];          a0.q[1] = As4[m0 * 4 + 2 + hi];
    a1.q[0] = As4[(m0 + 16) * 4 + hi];   a1.q[1] = As4[(m0 + 16) * 4 + 2 + hi];
    // B lane: col n; K = {8*hi+0..7, 16+8*hi+0..7}
    b0.q[0] = Bs4[n0 * 4 + hi];      b0.q[1] = Bs4[n0 * 4 + 2 + hi];
    b1.q[0] = Bs4[(n0 + 16) * 4 + hi]; b1.q[1] = Bs4[(n0 + 16) * 4 + 2 + hi];

    accs[0][0] = __builtin_amdgcn_wmma_f32_16x16x32_bf16(false, a0.v, false, b0.v, (short)0, accs[0][0], false, false);
    accs[0][1] = __builtin_amdgcn_wmma_f32_16x16x32_bf16(false, a0.v, false, b1.v, (short)0, accs[0][1], false, false);
    accs[1][0] = __builtin_amdgcn_wmma_f32_16x16x32_bf16(false, a1.v, false, b0.v, (short)0, accs[1][0], false, false);
    accs[1][1] = __builtin_amdgcn_wmma_f32_16x16x32_bf16(false, a1.v, false, b1.v, (short)0, accs[1][1], false, false);
    __syncthreads();
  }

  // epilogue: C/D layout is VGPR r -> M = r + 8*(lane>=16), N = lane&15
  const int hi = lane >> 4;
  #pragma unroll
  for (int si = 0; si < 2; ++si) {
    #pragma unroll
    for (int sj = 0; sj < 2; ++sj) {
      const int n = tileN + wc * 32 + sj * 16 + (lane & 15);
      if (n < p.N) {
        const float bv = p.bias ? p.bias[n] : 0.f;
        const int mrow0 = tileM + wr * 32 + si * 16 + (hi << 3);
        #pragma unroll
        for (int r = 0; r < 8; ++r) {
          float v = accs[si][sj][r] * p.alpha + bv;
          if (p.flags & 1) v = fmaxf(v, 0.f);
          C[(ll)(mrow0 + r) * p.ldc + n] = v;
        }
      }
    }
  }
}

// ---------------------------------------------------------------- aux kernels
__global__ __launch_bounds__(128) void k_embed(const int* __restrict__ x,
                                               const float* __restrict__ emb,
                                               float* __restrict__ h) {
  using namespace fs;
  int b = blockIdx.y, s = blockIdx.x;
  int tok = x[b * S + s];
  const float* e = emb + (ll)tok * D;
  float* out = h + (ll)(b * S + s) * D;
  for (int d = threadIdx.x; d < D; d += blockDim.x) {
    int i = d >> 1;
    float ang = (float)s * powf(10000.f, -(2.f * (float)i) / (float)D);
    float pe = (d & 1) ? cosf(ang) : sinf(ang);
    out[d] = e[d] + pe;
  }
}

// wave-per-row softmax over trailing dim T (rows multiple of 8, T multiple of 4)
__global__ __launch_bounds__(256) void k_softmax(float* __restrict__ s, int T) {
  ll row = (ll)blockIdx.x * 8 + (threadIdx.x >> 5);
  int lane = threadIdx.x & 31;
  f32x4* r4 = (f32x4*)(s + row * T);
  const int n4 = T >> 2;
  float mx = -3.4e38f;
  for (int i = lane; i < n4; i += 32) {
    f32x4 v = r4[i];
    mx = fmaxf(mx, fmaxf(fmaxf(v[0], v[1]), fmaxf(v[2], v[3])));
  }
  mx = wmax(mx);
  float sum = 0.f;
  for (int i = lane; i < n4; i += 32) {
    f32x4 v = r4[i];
    v[0] = __expf(v[0] - mx); v[1] = __expf(v[1] - mx);
    v[2] = __expf(v[2] - mx); v[3] = __expf(v[3] - mx);
    r4[i] = v;
    sum += v[0] + v[1] + v[2] + v[3];
  }
  sum = wsum(sum);
  float inv = 1.f / sum;
  for (int i = lane; i < n4; i += 32) {
    f32x4 v = r4[i];
    v *= inv;
    r4[i] = v;
  }
}

// out = LN(xa [+ xb]) * g + b; wave per row of width Dm (rows mult. of 8, Dm mult. of 4)
__global__ __launch_bounds__(256) void k_layernorm(const float* __restrict__ xa,
                                                   const float* __restrict__ xb,
                                                   const float* __restrict__ g,
                                                   const float* __restrict__ bt,
                                                   float* __restrict__ out, int Dm) {
  ll row = (ll)blockIdx.x * 8 + (threadIdx.x >> 5);
  int lane = threadIdx.x & 31;
  const int n4 = Dm >> 2;
  const f32x4* pa = (const f32x4*)(xa + row * Dm);
  const f32x4* pb = xb ? (const f32x4*)(xb + row * Dm) : nullptr;
  float s = 0.f, ss = 0.f;
  for (int i = lane; i < n4; i += 32) {
    f32x4 v = pa[i];
    if (pb) v += pb[i];
    s  += v[0] + v[1] + v[2] + v[3];
    ss += v[0]*v[0] + v[1]*v[1] + v[2]*v[2] + v[3]*v[3];
  }
  s = wsum(s); ss = wsum(ss);
  float mean = s / (float)Dm;
  float var  = ss / (float)Dm - mean * mean;
  float rs   = rsqrtf(var + 1e-5f);
  const f32x4* g4  = (const f32x4*)g;
  const f32x4* bt4 = (const f32x4*)bt;
  f32x4* po = (f32x4*)(out + row * Dm);
  for (int i = lane; i < n4; i += 32) {
    f32x4 v = pa[i];
    if (pb) v += pb[i];
    po[i] = (v - mean) * rs * g4[i] + bt4[i];
  }
}

// torch Conv1d weight [Cout,Cin,3] -> tap-major [3*Cin, Cout]
__global__ __launch_bounds__(256) void k_repack(const float* __restrict__ w,
                                                float* __restrict__ wt,
                                                int Cout, int Cin) {
  int n = blockIdx.x * 256 + threadIdx.x;
  int total = Cout * Cin * 3;
  if (n >= total) return;
  int cout = n % Cout;
  int kk   = n / Cout;
  int kt = kk / Cin, ci = kk - kt * Cin;
  wt[n] = w[((ll)cout * Cin + ci) * 3 + kt];
}

// cumsum durations + searchsorted(right): idx[b,pos], mask[b,pos]
__global__ __launch_bounds__(128) void k_lrindex(const int* __restrict__ y,
                                                 int* __restrict__ idx,
                                                 float* __restrict__ maskf) {
  using namespace fs;
  __shared__ int ends[S];
  int b = blockIdx.x, t = threadIdx.x;
  ends[t] = y[b * S + t];
  __syncthreads();
  for (int off = 1; off < S; off <<= 1) {      // Hillis-Steele inclusive scan
    int v = (t >= off) ? ends[t - off] : 0;
    __syncthreads();
    ends[t] += v;
    __syncthreads();
  }
  int total = ends[S - 1];
  for (int pos = t; pos < LOUT; pos += 128) {
    int lo = 0, hi = S;
    while (lo < hi) { int mid = (lo + hi) >> 1; if (ends[mid] <= pos) lo = mid + 1; else hi = mid; }
    int ii = lo < S - 1 ? lo : S - 1;
    idx[b * LOUT + pos] = ii;
    maskf[b * LOUT + pos] = (pos < total) ? 1.f : 0.f;
  }
}

__global__ __launch_bounds__(128) void k_gather(const float* __restrict__ hs,
                                                const int* __restrict__ idx,
                                                const float* __restrict__ mk,
                                                float* __restrict__ dst) {
  using namespace fs;
  int b = blockIdx.y, pos = blockIdx.x;
  int ii  = idx[b * LOUT + pos];
  float m = mk[b * LOUT + pos];
  const f32x4* src = (const f32x4*)(hs + (ll)(b * S + ii) * D);
  f32x4* out = (f32x4*)(dst + (ll)(b * LOUT + pos) * D);
  for (int d = threadIdx.x; d < (D >> 2); d += 128) out[d] = src[d] * m;
}

// ---------------------------------------------------------------- host side
static void gemm(hipStream_t st,
                 const float* A, ll lda, ll sAb, ll sAh,
                 const float* Bm, ll ldb, ll sBb, ll sBh,
                 float* C, ll ldc, ll sCb, ll sCh,
                 int M, int N, int Kd, int Z, int Hh,
                 const float* bias, int flags, int convT, int convCin, float alpha) {
  int convTsh = 0;
  while (convT > (1 << convTsh) && convTsh < 30) ++convTsh;   // convT is a power of 2
  GemmP p{A, Bm, C, bias, lda, ldb, ldc, sAb, sAh, sBb, sBh, sCb, sCh,
          M, N, Kd, Hh, flags, convT, convTsh, convCin, alpha};
  dim3 g((N + fs::TN - 1) / fs::TN, M / fs::TM, Z);
  hipLaunchKernelGGL(k_gemm, g, dim3(256), 0, st, p);
}

static void fft_block(hipStream_t st, int T,
                      float* x, float* t1, float* t2, float* qkv, float* big, float* wrep,
                      const float* qkv_w, const float* qkv_b,
                      const float* out_w, const float* out_b,
                      const float* ln1_g, const float* ln1_b,
                      const float* c1_w, const float* c1_b,
                      const float* c2_w, const float* c2_b,
                      const float* ln2_g, const float* ln2_b) {
  using namespace fs;
  const int M = B * T;
  const float inv_sqrt_dh = 1.0f / sqrtf((float)DH);

  // qkv = x @ qkv_w + qkv_b                                [M, 3D]
  gemm(st, x, D, 0, 0, qkv_w, 3 * D, 0, 0, qkv, 3 * D, 0, 0,
       M, 3 * D, D, 1, 1, qkv_b, 0, 1, 0, 1.f);
  // scores[b,h] = Q Kt / sqrt(dh)  (Q,K strided views of qkv), transB mode
  gemm(st, qkv,      3 * D, (ll)T * 3 * D, DH,
           qkv + D,  3 * D, (ll)T * 3 * D, DH,
           big, T, (ll)H * T * T, (ll)T * T,
       T, T, DH, B * H, H, nullptr, 2, 1, 0, inv_sqrt_dh);
  hipLaunchKernelGGL(k_softmax, dim3((unsigned)((ll)B * H * T / 8)), dim3(256), 0, st, big, T);
  // t1[b, t, h*dh + d] = probs @ V
  gemm(st, big, T, (ll)H * T * T, (ll)T * T,
           qkv + 2 * D, 3 * D, (ll)T * 3 * D, DH,
           t1, D, (ll)T * D, DH,
       T, DH, T, B * H, H, nullptr, 0, 1, 0, 1.f);
  // t2 = t1 @ out_w + out_b
  gemm(st, t1, D, 0, 0, out_w, D, 0, 0, t2, D, 0, 0,
       M, D, D, 1, 1, out_b, 0, 1, 0, 1.f);
  // t1 = LN(x + t2)
  hipLaunchKernelGGL(k_layernorm, dim3(M / 8), dim3(256), 0, st, x, t2, ln1_g, ln1_b, t1, D);
  // conv1 (ReLU): big = relu(im2col(t1) @ Wt1 + c1_b)      [M, DF]
  hipLaunchKernelGGL(k_repack, dim3((3 * D * DF + 255) / 256), dim3(256), 0, st, c1_w, wrep, DF, D);
  gemm(st, t1, D, 0, 0, wrep, DF, 0, 0, big, DF, 0, 0,
       M, DF, 3 * D, 1, 1, c1_b, 1 | 4, T, D, 1.f);
  // conv2: t2 = im2col(big) @ Wt2 + c2_b                   [M, D]
  hipLaunchKernelGGL(k_repack, dim3((3 * DF * D + 255) / 256), dim3(256), 0, st, c2_w, wrep, D, DF);
  gemm(st, big, DF, 0, 0, wrep, D, 0, 0, t2, D, 0, 0,
       M, D, 3 * DF, 1, 1, c2_b, 4, T, DF, 1.f);
  // x = LN(t1 + t2)
  hipLaunchKernelGGL(k_layernorm, dim3(M / 8), dim3(256), 0, st, t1, t2, ln2_g, ln2_b, x, D);
}

extern "C" void kernel_launch(void* const* d_in, const int* in_sizes, int n_in,
                              void* d_out, int out_size, void* d_ws, size_t ws_size,
                              hipStream_t stream) {
  using namespace fs;
  (void)in_sizes; (void)n_in; (void)out_size; (void)ws_size;

  const int*   x_tok = (const int*)d_in[0];
  const int*   y_dur = (const int*)d_in[1];
  const float* emb   = (const float*)d_in[2];
  auto F = [&](int i) { return (const float*)d_in[i]; };
  // ph weights: 3..14, mel weights: 15..26, dp (27..36) is dead code, proj: 37..38
  const float* proj_w = F(37);
  const float* proj_b = F(38);

  // -------- workspace carve
  char* ws = (char*)d_ws;
  size_t off = 0;
  auto carve = [&](size_t bytes) {
    void* ptr = ws + off;
    off += (bytes + 255) & ~(size_t)255;
    return ptr;
  };
  const size_t szAct  = (size_t)B * LOUT * D * sizeof(float);
  const size_t szQkv  = (size_t)B * LOUT * 3 * D * sizeof(float);
  const size_t szBig  = (size_t)B * H * LOUT * LOUT * sizeof(float);  // >= conv hidden
  const size_t szWrep = (size_t)3 * D * DF * sizeof(float);
  float* bufA  = (float*)carve(szAct);
  float* bufB  = (float*)carve(szAct);
  float* bufC  = (float*)carve(szAct);
  float* qkv   = (float*)carve(szQkv);
  float* big   = (float*)carve(szBig);
  float* wrep  = (float*)carve(szWrep);
  int*   lridx = (int*)  carve((size_t)B * LOUT * sizeof(int));
  float* lrmsk = (float*)carve((size_t)B * LOUT * sizeof(float));

  // -------- phoneme encoder (T = S)
  hipLaunchKernelGGL(k_embed, dim3(S, B), dim3(128), 0, stream, x_tok, emb, bufA);
  for (int l = 0; l < L; ++l) {
    fft_block(stream, S, bufA, bufB, bufC, qkv, big, wrep,
              F(3)  + (size_t)l * D * 3 * D, F(4)  + (size_t)l * 3 * D,
              F(5)  + (size_t)l * D * D,     F(6)  + (size_t)l * D,
              F(7)  + (size_t)l * D,         F(8)  + (size_t)l * D,
              F(9)  + (size_t)l * DF * D * 3, F(10) + (size_t)l * DF,
              F(11) + (size_t)l * D * DF * 3, F(12) + (size_t)l * D,
              F(13) + (size_t)l * D,         F(14) + (size_t)l * D);
  }

  // -------- length regulation (teacher durations): bufA[B,S,D] -> bufB[B,LOUT,D]
  hipLaunchKernelGGL(k_lrindex, dim3(B), dim3(128), 0, stream, y_dur, lridx, lrmsk);
  hipLaunchKernelGGL(k_gather, dim3(LOUT, B), dim3(128), 0, stream, bufA, lridx, lrmsk, bufB);

  // -------- mel decoder (T = LOUT)
  for (int l = 0; l < L; ++l) {
    fft_block(stream, LOUT, bufB, bufA, bufC, qkv, big, wrep,
              F(15) + (size_t)l * D * 3 * D, F(16) + (size_t)l * 3 * D,
              F(17) + (size_t)l * D * D,     F(18) + (size_t)l * D,
              F(19) + (size_t)l * D,         F(20) + (size_t)l * D,
              F(21) + (size_t)l * DF * D * 3, F(22) + (size_t)l * DF,
              F(23) + (size_t)l * D * DF * 3, F(24) + (size_t)l * D,
              F(25) + (size_t)l * D,         F(26) + (size_t)l * D);
  }

  // -------- final projection: out = bufB @ proj_w + proj_b   [B*LOUT, NM]
  gemm(stream, bufB, D, 0, 0, proj_w, NM, 0, 0, (float*)d_out, NM, 0, 0,
       B * LOUT, NM, D, 1, 1, proj_b, 0, 1, 0, 1.f);
}
